// SplineCoupling_70600672412076
// MI455X (gfx1250) — matrix operand order repaired and checked by reading, not compiled, not run on hardware
//
#include <hip/hip_runtime.h>

// ---------------------------------------------------------------------------
// SplineCoupling on MI455X (gfx1250): bf16 WMMA implicit-GEMM conv pipeline.
// ---------------------------------------------------------------------------

typedef __attribute__((ext_vector_type(16))) __bf16 v16bf;
typedef __attribute__((ext_vector_type(8)))  __bf16 v8bf;
typedef __attribute__((ext_vector_type(8)))  float  v8f;

#define BATCH 32
#define CIN   12
#define CHALF 6
#define HID   128
#define IMG   128
#define HW    (IMG * IMG)        // 16384
#define NPIX  (BATCH * HW)       // 524288
#define COUT  54
#define COUTP 64

// workspace layout (bytes)
static constexpr size_t OFF_H1     = 0;                                   // bf16 [NPIX][128]
static constexpr size_t SZ_H1      = (size_t)NPIX * HID * 2;              // 134217728
static constexpr size_t OFF_H2     = OFF_H1 + SZ_H1;                      // bf16 [NPIX][128]
static constexpr size_t SZ_H2      = (size_t)NPIX * HID * 2;
static constexpr size_t OFF_PARAMS = OFF_H2 + SZ_H2;                      // f32 [NPIX][54]
static constexpr size_t SZ_PARAMS  = (size_t)NPIX * COUT * 4;
static constexpr size_t OFF_W2BF   = OFF_PARAMS + SZ_PARAMS;              // bf16 [128][128]
static constexpr size_t SZ_W2BF    = (size_t)HID * HID * 2;
static constexpr size_t OFF_W3BF   = OFF_W2BF + SZ_W2BF;                  // bf16 [9][64][128]
static constexpr size_t OUT_ELEMS  = (size_t)BATCH * CIN * HW;            // 6291456

// --------------------------- prep kernels ----------------------------------

__global__ __launch_bounds__(256) void prep_w2_kernel(const float* __restrict__ w2,
                                                      __bf16* __restrict__ w2bf) {
    int i = blockIdx.x * 256 + threadIdx.x;            // 16384 elems, [O][I] == [N][K]
    w2bf[i] = (__bf16)w2[i];
}

__global__ __launch_bounds__(256) void prep_w3_kernel(const float* __restrict__ w3,
                                                      __bf16* __restrict__ w3bf) {
    int i = blockIdx.x * 256 + threadIdx.x;            // 9*64*128 = 73728
    int j = i / (COUTP * HID);                         // shift index (ky*3+kx)
    int o = (i / HID) % COUTP;                         // padded output channel
    int c = i % HID;                                   // input channel (K)
    float v = (o < COUT) ? w3[((size_t)o * HID + c) * 9 + j] : 0.f;
    w3bf[i] = (__bf16)v;
}

__global__ void init_obj_kernel(const float* __restrict__ objv, float* __restrict__ obj_out) {
    obj_out[threadIdx.x] = objv[threadIdx.x];
}

// --------------------------- conv1: 3x3, 6 -> 128, ReLU --------------------
// scalar f32 (minor FLOP share); output bf16 channels-last for the WMMA GEMMs.

__global__ __launch_bounds__(256)
void conv1_kernel(const float* __restrict__ x, const float* __restrict__ w1,
                  const float* __restrict__ b1, __bf16* __restrict__ h1) {
    __shared__ float sw[HID * CHALF * 9];              // 6912 floats = 27 KB
    for (int i = threadIdx.x; i < HID * CHALF * 9; i += 256) sw[i] = w1[i];
    __syncthreads();

    int pix = blockIdx.x * 2 + (threadIdx.x >> 7);     // 2 pixels per block
    int oc  = threadIdx.x & 127;
    int b   = pix / HW;
    int p   = pix % HW;
    int y   = p / IMG;
    int xx  = p % IMG;
    const float* xb = x + (size_t)b * CIN * HW;        // identity = channels 0..5
    const float* wp = sw + oc * (CHALF * 9);

    float acc = b1[oc];
#pragma unroll
    for (int c = 0; c < CHALF; ++c) {
#pragma unroll
        for (int ky = 0; ky < 3; ++ky) {
            int yy = y + ky - 1;
#pragma unroll
            for (int kx = 0; kx < 3; ++kx) {
                int xs = xx + kx - 1;
                float v = (yy >= 0 && yy < IMG && xs >= 0 && xs < IMG)
                              ? xb[(size_t)c * HW + yy * IMG + xs] : 0.f;
                acc = fmaf(v, wp[(c * 3 + ky) * 3 + kx], acc);
            }
        }
    }
    acc = fmaxf(acc, 0.f);
    h1[(size_t)pix * HID + oc] = (__bf16)acc;
}

// --------------------------- conv2: 1x1 GEMM, 128 -> 128, ReLU -------------
// Per wave: one 16-pixel M-tile x 8 N-tiles; K = 128 in 4 chunks of 32.

__global__ __launch_bounds__(128)
void conv2_kernel(const __bf16* __restrict__ h1, const __bf16* __restrict__ w2bf,
                  const float* __restrict__ b2, __bf16* __restrict__ h2) {
    __shared__ __bf16 sw[HID * HID];                   // 32 KB, [N][K]
    for (int i = threadIdx.x; i < (HID * HID) / 8; i += 128)
        ((v8bf*)sw)[i] = ((const v8bf*)w2bf)[i];
    __syncthreads();

    int lane  = threadIdx.x & 31;
    int wave  = threadIdx.x >> 5;
    int mbase = (blockIdx.x * 4 + wave) * 16;
    int hf    = lane >> 4;                             // K-half select
    int r     = lane & 15;                             // row (A) / col (B,D)

    v8f acc[8];
#pragma unroll
    for (int nt = 0; nt < 8; ++nt)
#pragma unroll
        for (int v = 0; v < 8; ++v) acc[nt][v] = 0.f;

    const __bf16* aRow = h1 + (size_t)(mbase + r) * HID + hf * 8;
#pragma unroll
    for (int kc = 0; kc < 4; ++kc) {
        int k0 = kc * 32;
        v8bf alo = *(const v8bf*)(aRow + k0);          // K 0-7 / 8-15 (per half)
        v8bf ahi = *(const v8bf*)(aRow + k0 + 16);     // K 16-23 / 24-31
        v16bf a;
#pragma unroll
        for (int i = 0; i < 8; ++i) { a[i] = alo[i]; a[8 + i] = ahi[i]; }
#pragma unroll
        for (int nt = 0; nt < 8; ++nt) {
            const __bf16* bp = sw + (size_t)(nt * 16 + r) * HID + k0 + hf * 8;
            v8bf blo = *(const v8bf*)bp;
            v8bf bhi = *(const v8bf*)(bp + 16);
            v16bf bm;
#pragma unroll
            for (int i = 0; i < 8; ++i) { bm[i] = blo[i]; bm[8 + i] = bhi[i]; }
            acc[nt] = __builtin_amdgcn_wmma_f32_16x16x32_bf16(
                false, a, false, bm, (short)0, acc[nt], false, false);
        }
    }
    // epilogue: bias + ReLU -> bf16 channels-last
#pragma unroll
    for (int nt = 0; nt < 8; ++nt) {
        int n = nt * 16 + r;
        float bias = b2[n];
#pragma unroll
        for (int v = 0; v < 8; ++v) {
            int m = hf * 8 + v;
            float val = fmaxf(acc[nt][v] + bias, 0.f);
            h2[(size_t)(mbase + m) * HID + n] = (__bf16)val;
        }
    }
}

// --------------------------- conv3: 3x3 as 9 shifted GEMMs, 128 -> 54 ------
// Per wave: 16-pixel M-tile (one image row) x 4 N-tiles (N padded to 64);
// per shift j: 16 KB weight slab staged in LDS, 4 K-chunks x 4 N-tiles WMMA.

__global__ __launch_bounds__(256)
void conv3_kernel(const __bf16* __restrict__ h2, const __bf16* __restrict__ w3bf,
                  const float* __restrict__ b3, float* __restrict__ params) {
    __shared__ __bf16 sw[COUTP * HID];                 // 16 KB slab for one shift
    int lane  = threadIdx.x & 31;
    int wave  = threadIdx.x >> 5;
    int mbase = (blockIdx.x * 8 + wave) * 16;
    int b  = mbase / HW;
    int p  = mbase % HW;
    int y  = p / IMG;
    int x0 = p % IMG;                                  // tile never straddles rows (128 % 16 == 0)
    int hf = lane >> 4;
    int r  = lane & 15;

    v8bf zero8;
#pragma unroll
    for (int i = 0; i < 8; ++i) zero8[i] = (__bf16)0.f;

    v8f acc[4];
#pragma unroll
    for (int nt = 0; nt < 4; ++nt)
#pragma unroll
        for (int v = 0; v < 8; ++v) acc[nt][v] = 0.f;

    for (int j = 0; j < 9; ++j) {
        __syncthreads();
        for (int i = threadIdx.x; i < (COUTP * HID) / 8; i += 256)
            ((v8bf*)sw)[i] = ((const v8bf*)(w3bf + (size_t)j * COUTP * HID))[i];
        __syncthreads();

        int dy = j / 3 - 1, dx = j % 3 - 1;
        int yy = y + dy;
        if (yy < 0 || yy >= IMG) continue;             // wave-uniform; zero contribution

        int xs = x0 + r + dx;                          // shifted input x for this lane's row
        bool valid = (xs >= 0 && xs < IMG);
        const __bf16* aRow = h2 + ((size_t)b * HW + (size_t)yy * IMG + xs) * HID + hf * 8;
        __builtin_prefetch(valid ? (const void*)aRow : (const void*)h2, 0, 0);

#pragma unroll
        for (int kc = 0; kc < 4; ++kc) {
            int k0 = kc * 32;
            v8bf alo = zero8, ahi = zero8;
            if (valid) {                               // lane-level zero pad in x
                alo = *(const v8bf*)(aRow + k0);
                ahi = *(const v8bf*)(aRow + k0 + 16);
            }
            v16bf a;
#pragma unroll
            for (int i = 0; i < 8; ++i) { a[i] = alo[i]; a[8 + i] = ahi[i]; }
#pragma unroll
            for (int nt = 0; nt < 4; ++nt) {
                const __bf16* bp = sw + (size_t)(nt * 16 + r) * HID + k0 + hf * 8;
                v8bf blo = *(const v8bf*)bp;
                v8bf bhi = *(const v8bf*)(bp + 16);
                v16bf bm;
#pragma unroll
                for (int i = 0; i < 8; ++i) { bm[i] = blo[i]; bm[8 + i] = bhi[i]; }
                acc[nt] = __builtin_amdgcn_wmma_f32_16x16x32_bf16(
                    false, a, false, bm, (short)0, acc[nt], false, false);
            }
        }
    }
    // epilogue: bias, drop padded channels, f32 channels-last [pixel][54]
#pragma unroll
    for (int nt = 0; nt < 4; ++nt) {
        int n = nt * 16 + r;
        if (n < COUT) {
            float bias = b3[n];
#pragma unroll
            for (int v = 0; v < 8; ++v) {
                int m = hf * 8 + v;
                params[(size_t)(mbase + m) * COUT + n] = acc[nt][v] + bias;
            }
        }
    }
}

// --------------------------- spline + identity copy + objective ------------

__global__ __launch_bounds__(256)
void spline_kernel(const float* __restrict__ x, const float* __restrict__ params,
                   float* __restrict__ out, float* __restrict__ obj) {
    __shared__ float red[256];
    int gtid = blockIdx.x * 256 + threadIdx.x;         // 32*6*16384 elements
    int b    = gtid / (CHALF * HW);                    // whole block shares one b
    int rem  = gtid % (CHALF * HW);
    int c    = rem / HW;
    int pix  = rem % HW;

    size_t idI = ((size_t)b * CIN + c) * HW + pix;
    size_t idT = ((size_t)b * CIN + CHALF + c) * HW + pix;
    out[idI] = x[idI];                                 // identity half passthrough
    float xt = x[idT];

    const float* pp = params + ((size_t)b * HW + pix) * COUT + c * 9;
    float uw[5], uh[4];
#pragma unroll
    for (int i = 0; i < 5; ++i) uw[i] = pp[i];
#pragma unroll
    for (int i = 0; i < 4; ++i) uh[i] = pp[5 + i];

    // softmax widths
    float mx = uw[0];
#pragma unroll
    for (int i = 1; i < 5; ++i) mx = fmaxf(mx, uw[i]);
    float ew[5], s = 0.f;
#pragma unroll
    for (int i = 0; i < 5; ++i) { ew[i] = __expf(uw[i] - mx); s += ew[i]; }
    float inv = 1.f / s;
    float w_[5];
#pragma unroll
    for (int i = 0; i < 5; ++i) w_[i] = 0.001f + 0.995f * ew[i] * inv;

    float e_[4];
#pragma unroll
    for (int i = 0; i < 4; ++i) e_[i] = __expf(uh[i]);

    float fw = 0.5f * w_[0], lw = 0.5f * w_[4];
    float num = 0.5f * fw * e_[0] + 0.5f * lw * e_[3]
              + 0.5f * (e_[0] + e_[1]) * w_[1]
              + 0.5f * (e_[1] + e_[2]) * w_[2]
              + 0.5f * (e_[2] + e_[3]) * w_[3];
    float c0 = num / (1.f - 0.5f * fw - 0.5f * lw);

    float h_[6] = {c0, e_[0], e_[1], e_[2], e_[3], c0};
    float area = 0.f;
#pragma unroll
    for (int i = 0; i < 5; ++i) area += 0.5f * (h_[i] + h_[i + 1]) * w_[i];
    float ia = 1.f / area;
#pragma unroll
    for (int i = 0; i < 6; ++i) h_[i] = 0.001f + 0.999f * h_[i] * ia;

    float d[5];
#pragma unroll
    for (int i = 0; i < 5; ++i) d[i] = 0.5f * (h_[i] + h_[i + 1]) * w_[i];
    float cdf[6], loc[6];
    cdf[0] = 0.f; cdf[1] = d[0]; cdf[2] = cdf[1] + d[1];
    cdf[3] = cdf[2] + d[2]; cdf[4] = cdf[3] + d[3]; cdf[5] = 1.f;
    loc[0] = 0.f; loc[1] = w_[0]; loc[2] = loc[1] + w_[1];
    loc[3] = loc[2] + w_[2]; loc[4] = loc[3] + w_[3]; loc[5] = 1.f;

    bool inside = (xt >= -1.f) && (xt <= 1.f);
    float xc  = fminf(fmaxf(xt, -1.f), 1.f);
    float inp = (xc + 1.f) * 0.5f;
    int idx = -1;
#pragma unroll
    for (int i = 0; i < 6; ++i) idx += (inp >= loc[i]) ? 1 : 0;
    idx = min(max(idx, 0), 4);

    float locS = loc[0], cdfS = cdf[0], hl = h_[0], hr = h_[1], wS = w_[0];
#pragma unroll
    for (int i = 1; i < 5; ++i)
        if (idx == i) { locS = loc[i]; cdfS = cdf[i]; hl = h_[i]; hr = h_[i + 1]; wS = w_[i]; }

    float alpha = (inp - locS) / wS;
    float aq = 0.5f * (hr - hl) * wS, bq = hl * wS;
    float o = aq * alpha * alpha + bq * alpha + cdfS;
    o = fminf(fmaxf(o, 0.f), 1.f) * 2.f - 1.f;
    float lad = __logf(alpha * (hr - hl) + hl);

    out[idT] = inside ? o : xt;
    lad = inside ? lad : 0.f;

    // block reduction of log-abs-det into per-batch objective
    red[threadIdx.x] = lad;
    __syncthreads();
#pragma unroll
    for (int stp = 128; stp > 0; stp >>= 1) {
        if (threadIdx.x < stp) red[threadIdx.x] += red[threadIdx.x + stp];
        __syncthreads();
    }
    if (threadIdx.x == 0) atomicAdd(&obj[b], red[0]);
}

// --------------------------- launch ----------------------------------------

extern "C" void kernel_launch(void* const* d_in, const int* in_sizes, int n_in,
                              void* d_out, int out_size, void* d_ws, size_t ws_size,
                              hipStream_t stream) {
    const float* x    = (const float*)d_in[0];
    const float* objv = (const float*)d_in[1];
    const float* w1   = (const float*)d_in[2];
    const float* b1   = (const float*)d_in[3];
    const float* w2   = (const float*)d_in[4];
    const float* b2   = (const float*)d_in[5];
    const float* w3   = (const float*)d_in[6];
    const float* b3   = (const float*)d_in[7];
    float* out = (float*)d_out;
    char*  ws  = (char*)d_ws;

    __bf16* h1      = (__bf16*)(ws + OFF_H1);
    __bf16* h2      = (__bf16*)(ws + OFF_H2);
    float*  paramsB = (float*)(ws + OFF_PARAMS);
    __bf16* w2bf    = (__bf16*)(ws + OFF_W2BF);
    __bf16* w3bf    = (__bf16*)(ws + OFF_W3BF);
    float*  obj_out = out + OUT_ELEMS;

    prep_w2_kernel<<<(HID * HID) / 256, 256, 0, stream>>>(w2, w2bf);
    prep_w3_kernel<<<(9 * COUTP * HID) / 256, 256, 0, stream>>>(w3, w3bf);
    init_obj_kernel<<<1, BATCH, 0, stream>>>(objv, obj_out);

    conv1_kernel<<<NPIX / 2, 256, 0, stream>>>(x, w1, b1, h1);
    conv2_kernel<<<NPIX / (16 * 4), 128, 0, stream>>>(h1, w2bf, b2, h2);
    conv3_kernel<<<NPIX / (16 * 8), 256, 0, stream>>>(h2, w3bf, b3, paramsB);
    spline_kernel<<<(BATCH * CHALF * HW) / 256, 256, 0, stream>>>(x, paramsB, out, obj_out);
}
